// SparseMOE_12060268167904
// MI455X (gfx1250) — matrix operand order repaired and checked by reading, not compiled, not run on hardware
//
#include <hip/hip_runtime.h>

typedef __attribute__((ext_vector_type(16))) __bf16 v16bf;
typedef __attribute__((ext_vector_type(8)))  float  v8f;

#define N_TOKENS  65536
#define D_IN      1024
#define D_OUT     1024
#define N_EXPERTS 64

// workspace layout (float units unless noted):
//   s      : [0, 65536)            expert-aggregated tokens  [E][D_IN]
//   sum_w  : [65536, 65600)        per-expert gate-weight sums
//   tot    : [65600, 66624)        final 1024-wide reduction
//   WgT    : byte offset 266496, 65536 bf16 = 128 KB, Wg transposed [E][D_IN]
#define WS_ZERO_FLOATS 66624
#define WS_WGT_BYTE_OFF 266496

__global__ void zero_ws_kernel(float* __restrict__ ws) {
  int i = blockIdx.x * blockDim.x + threadIdx.x;
  if (i < WS_ZERO_FLOATS) ws[i] = 0.0f;
}

__global__ void make_wgt_kernel(const float* __restrict__ Wg, __bf16* __restrict__ WgT) {
  int i = blockIdx.x * blockDim.x + threadIdx.x;   // i = e*1024 + k
  int e = i >> 10;
  int k = i & 1023;
  WgT[i] = (__bf16)Wg[k * N_EXPERTS + e];
}

// 256 threads (8 waves). LDS: s_local[64*1024] + sum_w[64] + bg[64] + 8*1024 logit scratch
// = 295,424 B of the 320 KB WGP LDS -> one workgroup per WGP.
__global__ void __launch_bounds__(256, 1) moe_gate_scatter_kernel(
    const float* __restrict__ x, const __bf16* __restrict__ WgT,
    const float* __restrict__ bg, float* __restrict__ s_g, float* __restrict__ sumw_g)
{
  extern __shared__ float smem[];
  float* s_l      = smem;            // 65536
  float* sumw_l   = smem + 65536;    // 64
  float* bg_l     = smem + 65600;    // 64
  float* lscratch = smem + 65664;    // 8 waves * 1024 (16 tokens x 64 experts)

  const int tid    = threadIdx.x;
  const int wave   = tid >> 5;
  const int lane   = tid & 31;
  const int lane16 = lane & 15;
  const int hs     = lane >> 4;      // half-wave select

  for (int i = tid; i < 65536; i += 256) s_l[i] = 0.0f;
  if (tid < N_EXPERTS) { sumw_l[tid] = 0.0f; bg_l[tid] = bg[tid]; }
  __syncthreads();

  float* mylog = lscratch + wave * 1024;
  const int wg_tok0 = blockIdx.x * 256;

  for (int blk = wave; blk < 16; blk += 8) {
    const int tok0 = wg_tok0 + blk * 16;
    v8f acc0 = {}, acc1 = {}, acc2 = {}, acc3 = {};

    // A fragment: this lane owns row (token) lane16; contiguous k order used here:
    // lanes 0-15: K = kk+0..15; lanes 16-31: K = kk+16..31 (matches B).
    const float* arow = x + (size_t)(tok0 + lane16) * D_IN;

    for (int kk = 0; kk < D_IN; kk += 32) {
      const float4* g0 = (const float4*)(arow + kk + hs * 16);
      const float4* g1 = (const float4*)(arow + kk + 8 + hs * 16);
      float4 f0 = g0[0], f1 = g0[1], f2 = g1[0], f3 = g1[1];
      v16bf a;
      a[0]=(__bf16)f0.x;  a[1]=(__bf16)f0.y;  a[2]=(__bf16)f0.z;  a[3]=(__bf16)f0.w;
      a[4]=(__bf16)f1.x;  a[5]=(__bf16)f1.y;  a[6]=(__bf16)f1.z;  a[7]=(__bf16)f1.w;
      a[8]=(__bf16)f2.x;  a[9]=(__bf16)f2.y;  a[10]=(__bf16)f2.z; a[11]=(__bf16)f2.w;
      a[12]=(__bf16)f3.x; a[13]=(__bf16)f3.y; a[14]=(__bf16)f3.z; a[15]=(__bf16)f3.w;

      // B fragment: lane owns expert column lane16; lanes 0-15 K=kk..kk+15,
      // lanes 16-31 K=kk+16..kk+31 -> 32 contiguous bytes of WgT[e][*].
      const __bf16* bbase = WgT + kk + hs * 16;
      v16bf b0 = *(const v16bf*)(bbase + (size_t)( 0 + lane16) * D_IN);
      v16bf b1 = *(const v16bf*)(bbase + (size_t)(16 + lane16) * D_IN);
      v16bf b2 = *(const v16bf*)(bbase + (size_t)(32 + lane16) * D_IN);
      v16bf b3 = *(const v16bf*)(bbase + (size_t)(48 + lane16) * D_IN);

      acc0 = __builtin_amdgcn_wmma_f32_16x16x32_bf16(false, a, false, b0, (short)0, acc0, false, false);
      acc1 = __builtin_amdgcn_wmma_f32_16x16x32_bf16(false, a, false, b1, (short)0, acc1, false, false);
      acc2 = __builtin_amdgcn_wmma_f32_16x16x32_bf16(false, a, false, b2, (short)0, acc2, false, false);
      acc3 = __builtin_amdgcn_wmma_f32_16x16x32_bf16(false, a, false, b3, (short)0, acc3, false, false);
    }

    // Spill 16x64 logits: C/D layout -> row = j + 8*hs, col = tile*16 + lane16
    #pragma unroll
    for (int j = 0; j < 8; ++j) {
      int row = j + 8 * hs;
      mylog[row * 64 +  0 + lane16] = acc0[j];
      mylog[row * 64 + 16 + lane16] = acc1[j];
      mylog[row * 64 + 32 + lane16] = acc2[j];
      mylog[row * 64 + 48 + lane16] = acc3[j];
    }
    asm volatile("s_wait_dscnt 0" ::: "memory");   // wave-private scratch; no WG barrier needed

    // Per-lane top-2 + softmax over the two selected logits
    int i0 = 0, i1 = 0; float w0 = 0.0f, w1 = 0.0f;
    if (lane < 16) {
      const float* lrow = mylog + lane16 * 64;
      float best = -3.4e38f, second = -3.4e38f; int bi = 0, si = 0;
      for (int e = 0; e < N_EXPERTS; ++e) {
        float v = lrow[e] + bg_l[e];
        if (v > best)        { second = best; si = bi; best = v; bi = e; }
        else if (v > second) { second = v; si = e; }
      }
      float e2  = __expf(second - best);
      float inv = 1.0f / (1.0f + e2);
      i0 = bi; w0 = inv;
      i1 = si; w1 = e2 * inv;
      atomicAdd(&sumw_l[i0], w0);
      atomicAdd(&sumw_l[i1], w1);
    }

    // Scatter w * x[token] into LDS expert accumulators (ds_add_f32)
    for (int t = 0; t < 16; ++t) {
      int   e0  = __builtin_amdgcn_readlane(i0, t);
      int   e1  = __builtin_amdgcn_readlane(i1, t);
      float ww0 = __uint_as_float(__builtin_amdgcn_readlane(__float_as_uint(w0), t));
      float ww1 = __uint_as_float(__builtin_amdgcn_readlane(__float_as_uint(w1), t));
      const float* xrow = x + (size_t)(tok0 + t) * D_IN;   // hot in cache from WMMA pass
      float* d0 = s_l + e0 * D_IN;
      float* d1 = s_l + e1 * D_IN;
      #pragma unroll 4
      for (int c = lane; c < D_IN; c += 32) {
        float v = xrow[c];
        atomicAdd(&d0[c], ww0 * v);
        atomicAdd(&d1[c], ww1 * v);
      }
    }
  }

  __syncthreads();
  for (int i = tid; i < 65536; i += 256) atomicAdd(&s_g[i], s_l[i]);
  if (tid < N_EXPERTS) atomicAdd(&sumw_g[tid], sumw_l[tid]);
}

// tot[o] = sum_r s[r] * We_flat[r][o] + sum_e sumw[e]*be[e][o]; streams We (268 MB) once.
__global__ void __launch_bounds__(256) moe_reduce_kernel(
    const float* __restrict__ s, const float* __restrict__ We,
    const float* __restrict__ sumw, const float* __restrict__ be,
    float* __restrict__ tot)
{
  const int tid = threadIdx.x;          // float4 column 0..255
  const int blk = blockIdx.x;           // 256 blocks, 256 rows each
  float4 acc = make_float4(0.f, 0.f, 0.f, 0.f);
  const int r0 = blk * 256;
  for (int r = r0; r < r0 + 256; ++r) {
    float  sv = s[r];                                        // uniform -> scalar load
    float4 wv = ((const float4*)(We + (size_t)r * D_OUT))[tid];
    acc.x += sv * wv.x; acc.y += sv * wv.y;
    acc.z += sv * wv.z; acc.w += sv * wv.w;
  }
  if (blk < N_EXPERTS) {
    float  sw = sumw[blk];
    float4 bv = ((const float4*)(be + (size_t)blk * D_OUT))[tid];
    acc.x += sw * bv.x; acc.y += sw * bv.y;
    acc.z += sw * bv.z; acc.w += sw * bv.w;
  }
  float* t = tot + tid * 4;
  atomicAdd(t + 0, acc.x); atomicAdd(t + 1, acc.y);
  atomicAdd(t + 2, acc.z); atomicAdd(t + 3, acc.w);
}

// out[n][:] = tot[:] for all n — 268 MB of float4 stores.
__global__ void __launch_bounds__(256) moe_broadcast_kernel(
    const float* __restrict__ tot, float* __restrict__ out)
{
  size_t gid = (size_t)blockIdx.x * blockDim.x + threadIdx.x;  // N*D/4 float4 slots
  int c4 = (int)(gid & (D_OUT / 4 - 1));
  ((float4*)out)[gid] = ((const float4*)tot)[c4];
}

extern "C" void kernel_launch(void* const* d_in, const int* in_sizes, int n_in,
                              void* d_out, int out_size, void* d_ws, size_t ws_size,
                              hipStream_t stream) {
  const float* x  = (const float*)d_in[0];   // [N, D_IN]
  const float* Wg = (const float*)d_in[1];   // [D_IN, E]
  const float* bg = (const float*)d_in[2];   // [E]
  const float* We = (const float*)d_in[3];   // [E, D_IN, D_OUT]
  const float* be = (const float*)d_in[4];   // [E, D_OUT]
  float* out = (float*)d_out;

  float*  ws     = (float*)d_ws;
  float*  s_g    = ws;
  float*  sumw_g = ws + 65536;
  float*  tot    = ws + 65600;
  __bf16* WgT    = (__bf16*)((char*)d_ws + WS_WGT_BYTE_OFF);

  zero_ws_kernel<<<(WS_ZERO_FLOATS + 255) / 256, 256, 0, stream>>>(ws);
  make_wgt_kernel<<<(N_EXPERTS * D_IN) / 256, 256, 0, stream>>>(Wg, WgT);

  size_t lds_bytes = (size_t)(65536 + 64 + 64 + 8 * 1024) * sizeof(float);  // 295,424 B
  moe_gate_scatter_kernel<<<N_TOKENS / 256, 256, lds_bytes, stream>>>(x, WgT, bg, s_g, sumw_g);

  moe_reduce_kernel<<<256, 256, 0, stream>>>(s_g, We, sumw_g, be, tot);

  moe_broadcast_kernel<<<(N_TOKENS * (D_OUT / 4)) / 256, 256, 0, stream>>>(tot, out);
}
